// QuantLinearINT8_3307124818666
// MI455X (gfx1250) — 8-bit matrix operand order repaired and checked by reading, compiled, not run on hardware
//
#include <hip/hip_runtime.h>
#include <stddef.h>
#include <stdint.h>


typedef __attribute__((ext_vector_type(8))) int          v8i;
typedef __attribute__((ext_vector_type(4))) int          v4i;
typedef __attribute__((ext_vector_type(2))) int          v2i;
typedef __attribute__((ext_vector_type(4))) float        v4f;
typedef __attribute__((ext_vector_type(4))) unsigned int u32x4;
typedef __attribute__((ext_vector_type(8))) unsigned int u32x8;

#define INF_  4096
#define OUTF_ 4096
#define SXINV 50.0f
#define SX    0.02f
#define BK    128
#define LDSTR 160

__global__ __launch_bounds__(256) void quant_x_kernel(
    const float* __restrict__ x, const float* __restrict__ balinv,
    int* __restrict__ xq, int ndw)
{
  int i = blockIdx.x * 256 + threadIdx.x;
  if (i >= ndw) return;
  v4f v = ((const v4f*)x)[i];
  int col = (i << 2) & (INF_ - 1);
  v4f b = *(const v4f*)(balinv + col);
  int q0 = (int)rintf(fminf(fmaxf(v.x * b.x * SXINV, -127.0f), 127.0f));
  int q1 = (int)rintf(fminf(fmaxf(v.y * b.y * SXINV, -127.0f), 127.0f));
  int q2 = (int)rintf(fminf(fmaxf(v.z * b.z * SXINV, -127.0f), 127.0f));
  int q3 = (int)rintf(fminf(fmaxf(v.w * b.w * SXINV, -127.0f), 127.0f));
  xq[i] = (q0 & 255) | ((q1 & 255) << 8) | ((q2 & 255) << 16) | ((q3 & 255) << 24);
}

__device__ __forceinline__ void tdm_load_tile(unsigned ldsAddr,
                                              const signed char* gp)
{
  unsigned long long ga = (unsigned long long)(uintptr_t)gp;
  u32x4 g0;
  g0[0] = 1u;
  g0[1] = ldsAddr;
  g0[2] = (unsigned)ga;
  g0[3] = ((unsigned)(ga >> 32) & 0x01FFFFFFu)
          | (2u << 30);
  u32x8 g1;
  g1[0] = (1u << 20)
        | (3u << 22)
        | (3u << 25);
  g1[1] = (unsigned)BK << 16;
  g1[2] = 128u << 16;
  g1[3] = (unsigned)BK << 16;
  g1[4] = 128u;
  g1[5] = (unsigned)INF_;
  g1[6] = 0u;
  g1[7] = 0u;
  asm volatile("tensor_load_to_lds %0, %1" :: "s"(g0), "s"(g1) : "memory");
}

__global__ __launch_bounds__(256) void gemm_i8_kernel(
    const signed char* __restrict__ xq,
    const signed char* __restrict__ w,
    const float* __restrict__ s_w,
    const float* __restrict__ bias,
    float* __restrict__ out,
    int T)
{
  __shared__ __align__(16) signed char As[2][128 * LDSTR];
  __shared__ __align__(16) signed char Bs[2][128 * LDSTR];

  const int tid   = threadIdx.x;
  const int lane  = tid & 31;
  const int wave  = tid >> 5;
  const int h     = lane >> 4;
  const int l16   = lane & 15;
  const int wm    = (wave >> 2) * 64;
  const int wn    = (wave & 3) * 32;
  const int tileM = blockIdx.x * 128;
  const int tileN = blockIdx.y * 128;

  v8i acc[4][2];
#pragma unroll
  for (int ms = 0; ms < 4; ++ms)
#pragma unroll
    for (int ns = 0; ns < 2; ++ns)
#pragma unroll
      for (int r = 0; r < 8; ++r) acc[ms][ns][r] = 0;

  const int NK = INF_ / BK;
  const signed char* Abase = xq + (size_t)tileM * INF_;
  const signed char* Bbase = w  + (size_t)tileN * INF_;

  const unsigned ldsA[2] = { (unsigned)(uintptr_t)&As[0][0],
                             (unsigned)(uintptr_t)&As[1][0] };
  const unsigned ldsB[2] = { (unsigned)(uintptr_t)&Bs[0][0],
                             (unsigned)(uintptr_t)&Bs[1][0] };

  const bool tdmWave = (wave == 0);

  if (tdmWave) {
    tdm_load_tile(ldsA[0], Abase);
    tdm_load_tile(ldsB[0], Bbase);
    __builtin_amdgcn_s_wait_tensorcnt(0);
  }
  __syncthreads();

  int buf = 0;
  for (int kt = 0; kt < NK; ++kt) {
    if (tdmWave && (kt + 1 < NK)) {
      const int kn = (kt + 1) * BK;
      tdm_load_tile(ldsA[buf ^ 1], Abase + kn);
      tdm_load_tile(ldsB[buf ^ 1], Bbase + kn);
    }

#pragma unroll
    for (int kk = 0; kk < 2; ++kk) {
      const int kb = kk * 80;

      v8i afrag[4];
#pragma unroll
      for (int ms = 0; ms < 4; ++ms) {
        const v2i* A2 = (const v2i*)&As[buf][(wm + ms * 16 + l16) * LDSTR + kb];
        v2i p0 = A2[2 * h], p1 = A2[2 * h + 1], p2 = A2[4 + 2 * h], p3 = A2[5 + 2 * h];
        afrag[ms][0] = p0.x; afrag[ms][1] = p0.y;
        afrag[ms][2] = p1.x; afrag[ms][3] = p1.y;
        afrag[ms][4] = p2.x; afrag[ms][5] = p2.y;
        afrag[ms][6] = p3.x; afrag[ms][7] = p3.y;
      }
      v8i bfrag[2];
#pragma unroll
      for (int ns = 0; ns < 2; ++ns) {
        const v4i* B4 = (const v4i*)&Bs[buf][(wn + ns * 16 + l16) * LDSTR + kb];
        v4i q0 = B4[h], q1 = B4[2 + h];
        bfrag[ns][0] = q0.x; bfrag[ns][1] = q0.y; bfrag[ns][2] = q0.z; bfrag[ns][3] = q0.w;
        bfrag[ns][4] = q1.x; bfrag[ns][5] = q1.y; bfrag[ns][6] = q1.z; bfrag[ns][7] = q1.w;
      }

#pragma unroll
      for (int ms = 0; ms < 4; ++ms)
#pragma unroll
        for (int ns = 0; ns < 2; ++ns)
          acc[ms][ns] = __builtin_amdgcn_wmma_i32_16x16x64_iu8(
              true, afrag[ms], true, bfrag[ns], acc[ms][ns], false, false);
    }

    if (tdmWave) __builtin_amdgcn_s_wait_tensorcnt(0);
    __syncthreads();
    buf ^= 1;
  }

#pragma unroll
  for (int ns = 0; ns < 2; ++ns) {
    const int col = tileN + wn + ns * 16 + l16;
    const float sc = SX * s_w[col];
    const float bv = bias[col];
#pragma unroll
    for (int ms = 0; ms < 4; ++ms) {
      const int rowBase = tileM + wm + ms * 16 + 8 * h;
      float* o = out + (size_t)rowBase * OUTF_ + col;
#pragma unroll
      for (int r = 0; r < 8; ++r)
        o[(size_t)r * OUTF_] = fmaf((float)acc[ms][ns][r], sc, bv);
    }
  }
}

extern "C" void kernel_launch(void* const* d_in, const int* in_sizes, int n_in,
                              void* d_out, int out_size, void* d_ws, size_t ws_size,
                              hipStream_t stream) {
  const float*       x      = (const float*)d_in[0];
  const signed char* w      = (const signed char*)d_in[1];
  const float*       s_w    = (const float*)d_in[2];
  const float*       balinv = (const float*)d_in[3];
  const float*       bias   = (const float*)d_in[4];
  float*             out    = (float*)d_out;

  const int T = in_sizes[0] / INF_;
  signed char* xq = (signed char*)d_ws;

  const int ndw = (T * INF_) / 4;
  quant_x_kernel<<<(ndw + 255) / 256, 256, 0, stream>>>(x, balinv, (int*)d_ws, ndw);

  dim3 grid(T / 128, OUTF_ / 128);
  gemm_i8_kernel<<<grid, 256, 0, stream>>>(xq, w, s_w, bias, out, T);
}
